// GraphRNN_81647328297095
// MI455X (gfx1250) — compile-verified, matrix operand order checked by reading, not run on hardware
//
#include <hip/hip_runtime.h>
#include <hip/hip_bf16.h>

typedef __attribute__((ext_vector_type(16))) _Float16 v16h;
typedef __attribute__((ext_vector_type(8)))  float    v8f;

#define D       256
#define THREE_D 768
#define MAT_ELEMS (768 * 256)    // one 3D x D weight matrix
#define KT      8                // 256 / 32 K-tiles
#define NT      48               // 768 / 16 N-tiles
#define NT_STRIDE 4096           // f16 elems between consecutive n-tiles (KT*32*16)
#define KT_STRIDE 512            // f16 elems between consecutive k-tiles (32*16)
#define BLOCK   512              // 16 wave32s per WG
#define NTW     3                // n-tiles per wave (16 * 3 = 48)

typedef const __attribute__((address_space(1))) _Float16 gf16;
typedef const __attribute__((address_space(1))) v16h     gv16h;

__device__ __forceinline__ float sigmoidf_(float x) {
    return 1.0f / (1.0f + __expf(-x));
}

// Offset of element (row, k) of a 16x256 matrix stored in WMMA A-fragment
// layout: [kt][lane][16] f16, lane = row + 16*hi.
__device__ __forceinline__ int afrag_off(int row, int k) {
    int kt = k >> 5;
    int w  = k & 31;
    int hi = (w >> 3) & 1;                 // [8,16) and [24,32) halves
    int j  = (w & 7) + ((w >= 16) ? 8 : 0);
    int lane = row + (hi << 4);
    return ((kt << 5) + lane) * 16 + j;
}

// Cooperative GEMM: OUT(16x768 fp32, LDS) = A(16x256) @ Wt + bias
// A: LDS f16, A-fragment layout. W: packed B-fragments in global.
// 512 threads (16 wave32s); wave w computes N-tiles [3w, 3w+3).
// Double-buffered across k-tiles: B tiles of kt+1 in flight during the
// 3 WMMAs of kt (low register pressure so the prefetch distance survives).
__device__ __forceinline__ void gemm16(const _Float16* __restrict__ Astage,
                                       const _Float16* Bp,
                                       const float* __restrict__ bias,
                                       float* __restrict__ OUT) {
    const int tid  = threadIdx.x;
    const int wave = tid >> 5;
    const int lane = tid & 31;
    const int r    = lane & 15;
    const int hi   = lane >> 4;
    const int n0   = wave * NTW;

    // Defeat LICM (prevents hoist+spill of loop-invariant weight loads across
    // the outer time-step loop), then restore the GLOBAL address space so the
    // loads lower to global_load_b128 (LOADcnt) instead of flat (LOADcnt+DScnt).
    unsigned long long p = (unsigned long long)Bp;
    asm volatile("" : "+s"(p));
    gf16* bl = (gf16*)p + ((size_t)(n0 * KT * 32 + lane) << 4);

    v8f acc[NTW];
#pragma unroll
    for (int i = 0; i < NTW; ++i) acc[i] = (v8f){};

    v16h a = *(const v16h*)(Astage + (lane << 4));
    v16h bc[NTW];
#pragma unroll
    for (int i = 0; i < NTW; ++i)
        bc[i] = *(gv16h*)(bl + i * NT_STRIDE);

#pragma unroll
    for (int kt = 0; kt < KT; ++kt) {
        v16h an;
        v16h bn[NTW];
        if (kt < KT - 1) {
            an = *(const v16h*)(Astage + ((((kt + 1) << 5) + lane) << 4));
#pragma unroll
            for (int i = 0; i < NTW; ++i)
                bn[i] = *(gv16h*)(bl + i * NT_STRIDE + (kt + 1) * KT_STRIDE);
        }
#pragma unroll
        for (int i = 0; i < NTW; ++i)
            acc[i] = __builtin_amdgcn_wmma_f32_16x16x32_f16(
                false, a, false, bc[i], (short)0, acc[i], false, false);
        if (kt < KT - 1) {
            a = an;
#pragma unroll
            for (int i = 0; i < NTW; ++i) bc[i] = bn[i];
        }
    }

    // D-layout: lane l holds OUT[v + 8*hi][nt*16 + r], v = 0..7
#pragma unroll
    for (int i = 0; i < NTW; ++i) {
        const int n = (n0 + i) * 16 + r;
        const float bv = bias[n];
#pragma unroll
        for (int v = 0; v < 8; ++v)
            OUT[(v + 8 * hi) * THREE_D + n] = acc[i][v] + bv;
    }
}

// ---------------- prep: pack weights fp32 -> f16 WMMA-B layout ----------------
// matrix order: 0:nWih0 1:nWih1 2:nWhh0 3:nWhh1 4:eWhh0 5:eWih1 6:eWhh1
__global__ void pack_weights_kernel(const float* __restrict__ nwih,
                                    const float* __restrict__ nwhh,
                                    const float* __restrict__ ewih,
                                    const float* __restrict__ ewhh,
                                    _Float16* __restrict__ pw) {
    size_t flat = (size_t)blockIdx.x * 256 + threadIdx.x;
    if (flat >= (size_t)7 * MAT_ELEMS) return;
    int m   = (int)(flat / MAT_ELEMS);
    int rem = (int)(flat % MAT_ELEMS);
    int nt   = rem >> 12;          // 8*32*16 = 4096 elems per n-tile
    int r2   = rem & 4095;
    int kt   = r2 >> 9;
    int r3   = r2 & 511;
    int lane = r3 >> 4;
    int j    = r3 & 15;
    int row  = nt * 16 + (lane & 15);
    int k    = kt * 32 + ((j < 8) ? j : j + 8) + (lane >> 4) * 8;
    const float* src;
    switch (m) {
        case 0: src = nwih;             break;
        case 1: src = nwih + MAT_ELEMS; break;
        case 2: src = nwhh;             break;
        case 3: src = nwhh + MAT_ELEMS; break;
        case 4: src = ewhh;             break;
        case 5: src = ewih + MAT_ELEMS; break;
        default: src = ewhh + MAT_ELEMS; break;
    }
    pw[flat] = (_Float16)src[row * D + k];
}

// rowsum of edge_w_ih[0] (scalar-input projection), collapsed regression head
__global__ void misc_prep_kernel(const float* __restrict__ ewih0,
                                 const float* __restrict__ rw1,
                                 const float* __restrict__ rb1,
                                 const float* __restrict__ rw2,
                                 const float* __restrict__ rb2,
                                 float* __restrict__ rowsum0,
                                 float* __restrict__ w_eff,
                                 float* __restrict__ b_eff) {
    int idx = blockIdx.x * 256 + threadIdx.x;
    if (idx < THREE_D) {
        float s = 0.0f;
        for (int d = 0; d < D; ++d) s += ewih0[idx * D + d];
        rowsum0[idx] = s;
    } else if (idx < THREE_D + D) {
        int d = idx - THREE_D;
        float s = 0.0f;
        for (int k = 0; k < 128; ++k) s += rw2[k] * rw1[k * D + d];
        w_eff[d] = s;
    } else if (idx == THREE_D + D) {
        float s = rb2[0];
        for (int k = 0; k < 128; ++k) s += rw2[k] * rb1[k];
        b_eff[0] = s;
    }
}

// ---------------- node-level GRU: 1 WG, both layers, M=4 padded to 16 --------
// LDS: A_X f16 frag [0,8192) B, A_H f16 frag [8192,16384) B,
//      XP fp32 [16384,65536) B, HP fp32 [65536,114688) B
__global__ void __launch_bounds__(BLOCK)
node_gru_kernel(const float* __restrict__ sos,
                const float* __restrict__ target,
                const _Float16* __restrict__ pw,
                const float* __restrict__ b_ih_all,
                const float* __restrict__ b_hh_all,
                float* __restrict__ out0,
                float* __restrict__ out1) {
    extern __shared__ __align__(16) char smem[];
    _Float16* AX = (_Float16*)(smem);
    _Float16* AH = (_Float16*)(smem + 8192);
    float* xp = (float*)(smem + 16384);
    float* hp = (float*)(smem + 65536);
    const int tid = threadIdx.x;

    for (int l = 0; l < 2; ++l) {
        const _Float16* Bih = pw + (size_t)l * MAT_ELEMS;
        const _Float16* Bhh = pw + (size_t)(2 + l) * MAT_ELEMS;
        const float* bih = b_ih_all + l * THREE_D;
        const float* bhh = b_hh_all + l * THREE_D;
        const float* lin = out0;
        float* lout = (l == 0) ? out0 : out1;

        // zero both f16 staging buffers (pad rows + h0 = 0)
        for (int i = tid; i < 4096; i += BLOCK) ((unsigned int*)smem)[i] = 0u;
        __syncthreads();

        for (int t = 0; t < 64; ++t) {
            for (int i = tid; i < 1024; i += BLOCK) {
                int b = i >> 8, d = i & 255;
                float v;
                if (l == 0)
                    v = (t == 0) ? sos[b * D + d]
                                 : target[((size_t)(t - 1) * 256 + d) * 4 + b];
                else
                    v = lin[(size_t)(t * 4 + b) * D + d];
                AX[afrag_off(b, d)] = (_Float16)v;
            }
            __syncthreads();
            gemm16(AX, Bih, bih, xp);
            gemm16(AH, Bhh, bhh, hp);
            __syncthreads();
            for (int i = tid; i < 1024; i += BLOCK) {
                int b = i >> 8, d = i & 255;
                int ao = afrag_off(b, d);
                float xr = xp[b * THREE_D + d];
                float xz = xp[b * THREE_D + 256 + d];
                float xn = xp[b * THREE_D + 512 + d];
                float hr = hp[b * THREE_D + d];
                float hz = hp[b * THREE_D + 256 + d];
                float hn = hp[b * THREE_D + 512 + d];
                float r = sigmoidf_(xr + hr);
                float z = sigmoidf_(xz + hz);
                float n = tanhf(xn + r * hn);
                float h = (float)AH[ao];
                float hnew = (1.0f - z) * n + z * h;
                AH[ao] = (_Float16)hnew;
                lout[(size_t)(t * 4 + b) * D + d] = hnew;
            }
            __syncthreads();
        }
        __syncthreads();
    }
}

// ------- edge-level GRU: 16 persistent WGs x 16 rows, fused L0+L1+regression --
// LDS bytes: A0 [0,8192) A1 [8192,16384) XP [16384,65536) HP [65536,114688)
//            ETOK [114688,114752) RS [114752,117824) BI [117824,120896)
__global__ void __launch_bounds__(BLOCK)
edge_gru_kernel(const float* __restrict__ target,
                const float* __restrict__ nodeout,
                const _Float16* __restrict__ pw,
                const float* __restrict__ eb_ih,
                const float* __restrict__ eb_hh,
                const float* __restrict__ rowsum0,
                const float* __restrict__ w_eff,
                const float* __restrict__ b_eff,
                float* __restrict__ out) {
    extern __shared__ __align__(16) char smem[];
    _Float16* A0 = (_Float16*)(smem);
    _Float16* A1 = (_Float16*)(smem + 8192);
    float* xp   = (float*)(smem + 16384);
    float* hp   = (float*)(smem + 65536);
    float* etok = (float*)(smem + 114688);
    float* rs   = (float*)(smem + 114752);
    float* bi   = (float*)(smem + 117824);
    const int tid = threadIdx.x;
    const int wg  = blockIdx.x;

    const _Float16* Bhh0 = pw + (size_t)4 * MAT_ELEMS;
    const _Float16* Bih1 = pw + (size_t)5 * MAT_ELEMS;
    const _Float16* Bhh1 = pw + (size_t)6 * MAT_ELEMS;
    const float* bhh0 = eb_hh;
    const float* bih1 = eb_ih + THREE_D;
    const float* bhh1 = eb_hh + THREE_D;

    // stage hidden init (both layers seeded with node output) + hot constants
    for (int i = tid; i < 4096; i += BLOCK) {
        int rr = i >> 8, d = i & 255;
        _Float16 v = (_Float16)nodeout[(size_t)(wg * 16) * D + i];
        int ao = afrag_off(rr, d);
        A0[ao] = v;
        A1[ao] = v;
    }
    for (int i = tid; i < THREE_D; i += BLOCK) {
        rs[i] = rowsum0[i];
        bi[i] = eb_ih[i];
    }
    __syncthreads();

    for (int e = 0; e < 256; ++e) {
        if (tid < 16) {
            int ng = wg * 16 + tid;
            int t = ng >> 2, b = ng & 3;
            etok[tid] = (e == 0) ? 1.0f
                                 : target[((size_t)t * 256 + (e - 1)) * 4 + b];
        }
        gemm16(A0, Bhh0, bhh0, hp);          // layer-0 recurrent
        __syncthreads();
        // layer-0 gates (input projection = scalar * rowsum + bias)
        for (int i = tid; i < 4096; i += BLOCK) {
            int rr = i >> 8, d = i & 255;
            int ao = afrag_off(rr, d);
            float s  = etok[rr];
            float xr = s * rs[d]       + bi[d];
            float xz = s * rs[256 + d] + bi[256 + d];
            float xn = s * rs[512 + d] + bi[512 + d];
            float hr = hp[rr * THREE_D + d];
            float hz = hp[rr * THREE_D + 256 + d];
            float hn = hp[rr * THREE_D + 512 + d];
            float r = sigmoidf_(xr + hr);
            float z = sigmoidf_(xz + hz);
            float n = tanhf(xn + r * hn);
            float h = (float)A0[ao];
            A0[ao] = (_Float16)((1.0f - z) * n + z * h);
        }
        __syncthreads();
        gemm16(A0, Bih1, bih1, xp);          // layer-1 input projection (fused)
        gemm16(A1, Bhh1, bhh1, hp);          // layer-1 recurrent
        __syncthreads();
        for (int i = tid; i < 4096; i += BLOCK) {
            int rr = i >> 8, d = i & 255;
            int ao = afrag_off(rr, d);
            float xr = xp[rr * THREE_D + d];
            float xz = xp[rr * THREE_D + 256 + d];
            float xn = xp[rr * THREE_D + 512 + d];
            float hr = hp[rr * THREE_D + d];
            float hz = hp[rr * THREE_D + 256 + d];
            float hn = hp[rr * THREE_D + 512 + d];
            float r = sigmoidf_(xr + hr);
            float z = sigmoidf_(xz + hz);
            float n = tanhf(xn + r * hn);
            float h = (float)A1[ao];
            A1[ao] = (_Float16)((1.0f - z) * n + z * h);
        }
        __syncthreads();
        // collapsed regression head: one dot per row
        if (tid < 16) {
            int ng = wg * 16 + tid;
            int t = ng >> 2, b = ng & 3;
            float sum = b_eff[0];
            for (int d = 0; d < D; ++d)
                sum += (float)A1[afrag_off(tid, d)] * w_eff[d];
            out[((size_t)t * 256 + e) * 4 + b] = sum;
        }
        __syncthreads();
    }
}

// ---------------------------------------------------------------------------
extern "C" void kernel_launch(void* const* d_in, const int* in_sizes, int n_in,
                              void* d_out, int out_size, void* d_ws, size_t ws_size,
                              hipStream_t stream) {
    (void)in_sizes; (void)n_in; (void)out_size; (void)ws_size;
    const float* sos    = (const float*)d_in[0];
    const float* target = (const float*)d_in[1];
    const float* nwih   = (const float*)d_in[2];
    const float* nwhh   = (const float*)d_in[3];
    const float* nbih   = (const float*)d_in[4];
    const float* nbhh   = (const float*)d_in[5];
    const float* ewih   = (const float*)d_in[6];
    const float* ewhh   = (const float*)d_in[7];
    const float* ebih   = (const float*)d_in[8];
    const float* ebhh   = (const float*)d_in[9];
    const float* rw1    = (const float*)d_in[10];
    const float* rb1    = (const float*)d_in[11];
    const float* rw2    = (const float*)d_in[12];
    const float* rb2    = (const float*)d_in[13];
    float* out = (float*)d_out;

    char* ws = (char*)d_ws;
    _Float16* pw    = (_Float16*)(ws + 0);          // 7 * 768*256 f16 = 2,752,512 B
    float* rowsum0  = (float*)(ws + 2752512);        // 768 f32
    float* w_eff    = (float*)(ws + 2755584);        // 256 f32
    float* b_eff    = (float*)(ws + 2756608);        // 1 f32 (padded)
    float* node_l0  = (float*)(ws + 2756864);        // 256*256 f32
    float* node_out = (float*)(ws + 3019008);        // 256*256 f32

    pack_weights_kernel<<<5376, 256, 0, stream>>>(nwih, nwhh, ewih, ewhh, pw);
    misc_prep_kernel<<<5, 256, 0, stream>>>(ewih, rw1, rb1, rw2, rb2,
                                            rowsum0, w_eff, b_eff);
    node_gru_kernel<<<1, BLOCK, 114688, stream>>>(sos, target, pw, nbih, nbhh,
                                                  node_l0, node_out);
    edge_gru_kernel<<<16, BLOCK, 120896, stream>>>(target, node_out, pw,
                                                   ebih, ebhh, rowsum0, w_eff,
                                                   b_eff, out);
}
